// SelectiveQuantizedTransformerBlock_47184510714572
// MI455X (gfx1250) — 16-bit and 8-bit matrix operand order repaired and checked by reading, compiled, not run on hardware
//
#include <hip/hip_runtime.h>
#include <hip/hip_bf16.h>
#include <math.h>

typedef _Float16 v8h  __attribute__((ext_vector_type(8)));
typedef _Float16 v16h __attribute__((ext_vector_type(16)));
typedef float    v8f  __attribute__((ext_vector_type(8)));
typedef int      v8i  __attribute__((ext_vector_type(8)));

static __device__ __forceinline__ v16h cat8h(v8h lo, v8h hi) {
  return __builtin_shufflevector(lo, hi, 0,1,2,3,4,5,6,7,8,9,10,11,12,13,14,15);
}
static __device__ __forceinline__ v16h cvt16(const float* __restrict__ p) {
  v16h r;
#pragma unroll
  for (int i = 0; i < 16; ++i) r[i] = (_Float16)p[i];
  return r;
}

#if defined(__AMDGCN__) && __has_builtin(__builtin_amdgcn_global_load_async_to_lds_b128) && __has_builtin(__builtin_amdgcn_s_wait_asynccnt)
#define USE_ASYNC_LDS 1
typedef int v4i_ __attribute__((vector_size(16)));
typedef __attribute__((address_space(1))) v4i_ g_v4i;
typedef __attribute__((address_space(3))) v4i_ l_v4i;
#else
#define USE_ASYNC_LDS 0
#endif

#define BB    2
#define NN    2048
#define DD    768
#define H3    2304
#define HID   3072
#define BN    4096
#define HEADS 12
#define HDIM  64


__global__ void zero_scales_kernel(float* sc) {
  if (threadIdx.x < 8) sc[threadIdx.x] = 0.0f;
}

__global__ __launch_bounds__(256)
void absmax_kernel(const float* __restrict__ src, size_t n, float* __restrict__ dst) {
  __shared__ float red[256];
  float am = 0.0f;
  for (size_t i = (size_t)blockIdx.x * 256 + threadIdx.x; i < n;
       i += (size_t)gridDim.x * 256)
    am = fmaxf(am, fabsf(src[i]));
  red[threadIdx.x] = am;
  __syncthreads();
  for (int s = 128; s > 0; s >>= 1) {
    if (threadIdx.x < s) red[threadIdx.x] = fmaxf(red[threadIdx.x], red[threadIdx.x + s]);
    __syncthreads();
  }
  if (threadIdx.x == 0)
    atomicMax((unsigned int*)dst, __float_as_uint(red[0]));
}

__global__ __launch_bounds__(256)
void quantize_kernel(const float* __restrict__ src, const float* __restrict__ amax,
                     signed char* __restrict__ dst, size_t n) {
  const float s   = fmaxf(amax[0] * (1.0f / 127.0f), 1e-8f);
  const float inv = 1.0f / s;
  for (size_t i = (size_t)blockIdx.x * 256 + threadIdx.x; i < n;
       i += (size_t)gridDim.x * 256) {
    int q = __float2int_rn(src[i] * inv);
    q = q > 127 ? 127 : (q < -128 ? -128 : q);
    dst[i] = (signed char)q;
  }
}

__global__ __launch_bounds__(256)
void layernorm_absmax_kernel(const float* __restrict__ x, const float* __restrict__ g,
                             const float* __restrict__ bta, float* __restrict__ out,
                             float* __restrict__ amax_out) {
  __shared__ float rs[256], rq[256];
  const int row = blockIdx.x;
  const float* xr = x + (size_t)row * DD;
  float v0 = xr[threadIdx.x], v1 = xr[threadIdx.x + 256], v2 = xr[threadIdx.x + 512];
  rs[threadIdx.x] = v0 + v1 + v2;
  rq[threadIdx.x] = v0 * v0 + v1 * v1 + v2 * v2;
  __syncthreads();
  for (int s = 128; s > 0; s >>= 1) {
    if (threadIdx.x < s) {
      rs[threadIdx.x] += rs[threadIdx.x + s];
      rq[threadIdx.x] += rq[threadIdx.x + s];
    }
    __syncthreads();
  }
  const float mu  = rs[0] * (1.0f / (float)DD);
  const float var = rq[0] * (1.0f / (float)DD) - mu * mu;
  const float inv = rsqrtf(var + 1e-5f);
  __syncthreads();
  float* orow = out + (size_t)row * DD;
  float am = 0.0f;
  float vv[3] = {v0, v1, v2};
#pragma unroll
  for (int i = 0; i < 3; ++i) {
    const int idx = threadIdx.x + 256 * i;
    const float o = (vv[i] - mu) * inv * g[idx] + bta[idx];
    orow[idx] = o;
    am = fmaxf(am, fabsf(o));
  }
  rs[threadIdx.x] = am;
  __syncthreads();
  for (int s = 128; s > 0; s >>= 1) {
    if (threadIdx.x < s) rs[threadIdx.x] = fmaxf(rs[threadIdx.x], rs[threadIdx.x + s]);
    __syncthreads();
  }
  if (threadIdx.x == 0)
    atomicMax((unsigned int*)amax_out, __float_as_uint(rs[0]));
}

__global__ __launch_bounds__(256)
void gemm_i8_wmma_kernel(const signed char* __restrict__ Aq,
                         const signed char* __restrict__ Wq,
                         float* __restrict__ Cout,
                         int M, int Nout, int K,
                         const float* __restrict__ amA, const float* __restrict__ amB,
                         const float* __restrict__ bias, const float* __restrict__ resid,
                         int do_gelu, float* __restrict__ amax_out) {
  __shared__ float red[256];
  const int wave = threadIdx.x >> 5;
  const int lane = threadIdx.x & 31;
  const int wm = wave & 1, wn = wave >> 1;
  const int l15 = lane & 15;
  const int mA = blockIdx.y * 32 + wm * 16 + l15;
  const int nB = blockIdx.x * 64 + wn * 16 + l15;
  const signed char* Arow = Aq + (size_t)mA * K;
  const signed char* Brow = Wq + (size_t)nB * K;
  const int a8  = (lane >> 4) << 3;
  const int b16 = (lane >> 4) << 4;

  v8i acc = {};
  for (int k0 = 0; k0 < K; k0 += 64) {
    __builtin_prefetch(Arow + k0 + 128, 0, 3);
    v8i af, bf;
#pragma unroll
    for (int c = 0; c < 4; ++c) {
      int2 t = *(const int2*)(Arow + k0 + b16 + 8 * (c & 1) + 32 * (c >> 1));
      af[2 * c] = t.x; af[2 * c + 1] = t.y;
    }
#pragma unroll
    for (int c = 0; c < 2; ++c) {
      int4 t = *(const int4*)(Brow + k0 + b16 + 32 * c);
      bf[4 * c] = t.x; bf[4 * c + 1] = t.y; bf[4 * c + 2] = t.z; bf[4 * c + 3] = t.w;
    }
    acc = __builtin_amdgcn_wmma_i32_16x16x64_iu8(true, af, true, bf, acc, false, false);
  }

  const float sA = fmaxf(amA[0] * (1.0f / 127.0f), 1e-8f);
  const float sB = fmaxf(amB[0] * (1.0f / 127.0f), 1e-8f);
  const float sc = sA * sB;
  const float bia = bias ? bias[nB] : 0.0f;
  const int mbase = blockIdx.y * 32 + wm * 16 + ((lane >> 4) << 3);
  float am = 0.0f;
#pragma unroll
  for (int r = 0; r < 8; ++r) {
    const int m = mbase + r;
    float y = (float)acc[r] * sc + bia;
    if (resid) y += resid[(size_t)m * Nout + nB];
    if (do_gelu) y = 0.5f * y * (1.0f + erff(y * 0.70710678118654752f));
    Cout[(size_t)m * Nout + nB] = y;
    am = fmaxf(am, fabsf(y));
  }
  if (amax_out) {
    red[threadIdx.x] = am;
    __syncthreads();
    for (int s = 128; s > 0; s >>= 1) {
      if (threadIdx.x < s) red[threadIdx.x] = fmaxf(red[threadIdx.x], red[threadIdx.x + s]);
      __syncthreads();
    }
    if (threadIdx.x == 0)
      atomicMax((unsigned int*)amax_out, __float_as_uint(red[0]));
  }
}

__global__ __launch_bounds__(128)
void attention_flash_kernel(const float* __restrict__ qkv, float* __restrict__ out,
                            float* __restrict__ amax_out) {
  __shared__ __align__(16) float    Kf32[32][64];
  __shared__ __align__(16) _Float16 Vt[64][32];
  __shared__ __align__(16) _Float16 Plds[4][16][32];
  __shared__ float red[128];

  const int bh = blockIdx.y;
  const int b = bh / HEADS, h = bh % HEADS;
  const int wave = threadIdx.x >> 5;
  const int lane = threadIdx.x & 31;
  const int l15 = lane & 15;
  const int qbase = blockIdx.x * 64 + wave * 16;
  const float scale = 0.125f;

  v16h qf[2];
  {
    const int m = qbase + l15;
    const size_t qoff = (((size_t)(b * NN + m)) * 3 + 0) * DD + (size_t)h * HDIM;
    const int koff = (lane >= 16) ? 16 : 0;
#pragma unroll
    for (int f = 0; f < 2; ++f) {
      const float* p0 = qkv + qoff + f * 32 + koff;
#pragma unroll
      for (int i = 0; i < 8; ++i) {
        qf[f][i]     = (_Float16)(p0[i] * scale);
        qf[f][i + 8] = (_Float16)(p0[i + 8] * scale);
      }
    }
  }

  float mrow[8], lrow[8];
  v8f O[4];
#pragma unroll
  for (int r = 0; r < 8; ++r) { mrow[r] = -3.0e38f; lrow[r] = 0.0f; }
#pragma unroll
  for (int t = 0; t < 4; ++t) O[t] = (v8f){};

  for (int jb = 0; jb < NN; jb += 32) {
#if USE_ASYNC_LDS
#pragma unroll
    for (int e = threadIdx.x; e < 32 * 16; e += 128) {
      const int key = e >> 4, c4 = e & 15;
      const float* gsrc =
          qkv + (((size_t)(b * NN + jb + key)) * 3 + 1) * DD + (size_t)h * HDIM + c4 * 4;
      __builtin_amdgcn_global_load_async_to_lds_b128(
          (g_v4i*)(size_t)gsrc,
          (l_v4i*)(unsigned int)(size_t)&Kf32[key][c4 * 4],
          0, 0);
    }
#else
    for (int e = threadIdx.x; e < 32 * 64; e += 128) {
      const int key = e >> 6, d = e & 63;
      Kf32[key][d] =
          qkv[(((size_t)(b * NN + jb + key)) * 3 + 1) * DD + (size_t)h * HDIM + d];
    }
#endif
    for (int e = threadIdx.x; e < 32 * 64; e += 128) {
      const int key = e >> 6, d = e & 63;
      Vt[d][key] = (_Float16)
          qkv[(((size_t)(b * NN + jb + key)) * 3 + 2) * DD + (size_t)h * HDIM + d];
    }
#if USE_ASYNC_LDS
    __builtin_amdgcn_s_wait_asynccnt(0);
#endif
    __syncthreads();

    v8f S0 = (v8f){}, S1 = (v8f){};
#pragma unroll
    for (int kc = 0; kc < 2; ++kc) {
      const int dbase = kc * 32 + ((lane < 16) ? 0 : 16);
      S0 = __builtin_amdgcn_wmma_f32_16x16x32_f16(false, qf[kc], false,
            cvt16(&Kf32[l15][dbase]), (short)0, S0, false, false);
      S1 = __builtin_amdgcn_wmma_f32_16x16x32_f16(false, qf[kc], false,
            cvt16(&Kf32[16 + l15][dbase]), (short)0, S1, false, false);
    }

#pragma unroll
    for (int r = 0; r < 8; ++r) {
      float s0 = S0[r], s1 = S1[r];
      float mx = fmaxf(s0, s1);
      mx = fmaxf(mx, __shfl_xor(mx, 1, 32));
      mx = fmaxf(mx, __shfl_xor(mx, 2, 32));
      mx = fmaxf(mx, __shfl_xor(mx, 4, 32));
      mx = fmaxf(mx, __shfl_xor(mx, 8, 32));
      const float nm = fmaxf(mrow[r], mx);
      const float corr = __expf(mrow[r] - nm);
      const float p0 = __expf(s0 - nm), p1 = __expf(s1 - nm);
      float rs = p0 + p1;
      rs += __shfl_xor(rs, 1, 32);
      rs += __shfl_xor(rs, 2, 32);
      rs += __shfl_xor(rs, 4, 32);
      rs += __shfl_xor(rs, 8, 32);
      lrow[r] = lrow[r] * corr + rs;
      mrow[r] = nm;
#pragma unroll
      for (int t = 0; t < 4; ++t) O[t][r] *= corr;
      const int mloc = ((lane >> 4) << 3) + r;
      Plds[wave][mloc][l15]      = (_Float16)p0;
      Plds[wave][mloc][16 + l15] = (_Float16)p1;
    }
    __syncthreads();

    v16h pf;
    {
      const int ko = (lane >= 16) ? 16 : 0;
      const v8h* p0 = (const v8h*)&Plds[wave][l15][ko];
      const v8h* p1 = (const v8h*)&Plds[wave][l15][ko + 8];
      pf = cat8h(p0[0], p1[0]);
    }
#pragma unroll
    for (int t = 0; t < 4; ++t) {
      const int dim = t * 16 + l15;
      const int kb = (lane < 16) ? 0 : 16;
      const v8h* p = (const v8h*)&Vt[dim][kb];
      O[t] = __builtin_amdgcn_wmma_f32_16x16x32_f16(false, pf, false,
              cat8h(p[0], p[1]), (short)0, O[t], false, false);
    }
    __syncthreads();
  }

  float am = 0.0f;
#pragma unroll
  for (int t = 0; t < 4; ++t)
#pragma unroll
    for (int r = 0; r < 8; ++r) {
      const int m = ((lane >> 4) << 3) + r;
      const float val = O[t][r] / lrow[r];
      out[((size_t)(b * NN) + qbase + m) * DD + (size_t)h * HDIM + t * 16 + l15] = val;
      am = fmaxf(am, fabsf(val));
    }
  red[threadIdx.x] = am;
  __syncthreads();
  for (int s = 64; s > 0; s >>= 1) {
    if (threadIdx.x < s) red[threadIdx.x] = fmaxf(red[threadIdx.x], red[threadIdx.x + s]);
    __syncthreads();
  }
  if (threadIdx.x == 0)
    atomicMax((unsigned int*)amax_out, __float_as_uint(red[0]));
}

#define WS_SC      ((size_t)0)
#define WS_LN1H    ((size_t)256)
#define WS_QKV     (WS_LN1H  + (size_t)BN * DD * 4)
#define WS_ATTNO   (WS_QKV   + (size_t)BN * H3 * 4)
#define WS_X1      (WS_ATTNO + (size_t)BN * DD * 4)
#define WS_LN2H    (WS_X1    + (size_t)BN * DD * 4)
#define WS_H1      (WS_LN2H  + (size_t)BN * DD * 4)
#define WS_AQ      (WS_H1    + (size_t)BN * HID * 4)
#define WS_WQ_QKV  (WS_AQ      + (size_t)BN * HID)
#define WS_WQ_PROJ (WS_WQ_QKV  + (size_t)H3 * DD)
#define WS_WQ_FC1  (WS_WQ_PROJ + (size_t)DD * DD)
#define WS_WQ_FC2  (WS_WQ_FC1  + (size_t)HID * DD)

extern "C" void kernel_launch(void* const* d_in, const int* in_sizes, int n_in,
                              void* d_out, int out_size, void* d_ws, size_t ws_size,
                              hipStream_t stream) {
  (void)in_sizes; (void)n_in; (void)out_size; (void)ws_size;
  const float* x      = (const float*)d_in[0];
  const float* ln1_g  = (const float*)d_in[1];
  const float* ln1_b  = (const float*)d_in[2];
  const float* ln2_g  = (const float*)d_in[3];
  const float* ln2_b  = (const float*)d_in[4];
  const float* qkv_w  = (const float*)d_in[5];
  const float* proj_w = (const float*)d_in[6];
  const float* fc1_w  = (const float*)d_in[7];
  const float* fc1_b  = (const float*)d_in[8];
  const float* fc2_w  = (const float*)d_in[9];
  const float* fc2_b  = (const float*)d_in[10];
  float* out = (float*)d_out;

  char* ws = (char*)d_ws;
  float* sc          = (float*)(ws + WS_SC);
  float* ln1h        = (float*)(ws + WS_LN1H);
  float* qkvbuf      = (float*)(ws + WS_QKV);
  float* attno       = (float*)(ws + WS_ATTNO);
  float* x1          = (float*)(ws + WS_X1);
  float* ln2h        = (float*)(ws + WS_LN2H);
  float* h1          = (float*)(ws + WS_H1);
  signed char* aq    = (signed char*)(ws + WS_AQ);
  signed char* wqqkv = (signed char*)(ws + WS_WQ_QKV);
  signed char* wqprj = (signed char*)(ws + WS_WQ_PROJ);
  signed char* wqf1  = (signed char*)(ws + WS_WQ_FC1);
  signed char* wqf2  = (signed char*)(ws + WS_WQ_FC2);

  zero_scales_kernel<<<1, 32, 0, stream>>>(sc);

  absmax_kernel<<<256, 256, 0, stream>>>(qkv_w,  (size_t)H3  * DD, sc + 1);
  absmax_kernel<<<256, 256, 0, stream>>>(proj_w, (size_t)DD  * DD, sc + 3);
  absmax_kernel<<<256, 256, 0, stream>>>(fc1_w,  (size_t)HID * DD, sc + 5);
  absmax_kernel<<<256, 256, 0, stream>>>(fc2_w,  (size_t)DD * HID, sc + 7);
  quantize_kernel<<<1024, 256, 0, stream>>>(qkv_w,  sc + 1, wqqkv, (size_t)H3  * DD);
  quantize_kernel<<<1024, 256, 0, stream>>>(proj_w, sc + 3, wqprj, (size_t)DD  * DD);
  quantize_kernel<<<1024, 256, 0, stream>>>(fc1_w,  sc + 5, wqf1,  (size_t)HID * DD);
  quantize_kernel<<<1024, 256, 0, stream>>>(fc2_w,  sc + 7, wqf2,  (size_t)DD * HID);

  layernorm_absmax_kernel<<<BN, 256, 0, stream>>>(x, ln1_g, ln1_b, ln1h, sc + 0);
  quantize_kernel<<<2048, 256, 0, stream>>>(ln1h, sc + 0, aq, (size_t)BN * DD);
  gemm_i8_wmma_kernel<<<dim3(H3 / 64, BN / 32), 256, 0, stream>>>(
      aq, wqqkv, qkvbuf, BN, H3, DD, sc + 0, sc + 1, nullptr, nullptr, 0, nullptr);
  attention_flash_kernel<<<dim3(NN / 64, BB * HEADS), 128, 0, stream>>>(
      qkvbuf, attno, sc + 2);
  quantize_kernel<<<2048, 256, 0, stream>>>(attno, sc + 2, aq, (size_t)BN * DD);
  gemm_i8_wmma_kernel<<<dim3(DD / 64, BN / 32), 256, 0, stream>>>(
      aq, wqprj, x1, BN, DD, DD, sc + 2, sc + 3, nullptr, x, 0, nullptr);

  layernorm_absmax_kernel<<<BN, 256, 0, stream>>>(x1, ln2_g, ln2_b, ln2h, sc + 4);
  quantize_kernel<<<2048, 256, 0, stream>>>(ln2h, sc + 4, aq, (size_t)BN * DD);
  gemm_i8_wmma_kernel<<<dim3(HID / 64, BN / 32), 256, 0, stream>>>(
      aq, wqf1, h1, BN, HID, DD, sc + 4, sc + 5, fc1_b, nullptr, 1, sc + 6);
  quantize_kernel<<<4096, 256, 0, stream>>>(h1, sc + 6, aq, (size_t)BN * HID);
  gemm_i8_wmma_kernel<<<dim3(DD / 64, BN / 32), 256, 0, stream>>>(
      aq, wqf2, out, BN, DD, HID, sc + 6, sc + 7, fc2_b, x1, 0, nullptr);
}
